// Decoder_14680198217759
// MI455X (gfx1250) — matrix operand order repaired and checked by reading, compiled, not run on hardware
//
#include <hip/hip_runtime.h>
#include <stdint.h>

// Problem dims (fixed by the reference)
#define BB   32
#define GG   512
#define NN   512
#define EE   128
#define HH   8
#define DD   16
#define HDIM 128

typedef __attribute__((ext_vector_type(8)))  float    v8f;
typedef __attribute__((ext_vector_type(8)))  uint32_t u32x8;
typedef __attribute__((ext_vector_type(16))) __bf16   bf16x16;

// ---- bf16 conversion helpers ----------------------------------------------
#if defined(__has_builtin)
#if __has_builtin(__builtin_amdgcn_cvt_pk_bf16_f32)
#define HAVE_PK_BF16 1
#endif
#endif

#ifdef HAVE_PK_BF16
typedef __attribute__((ext_vector_type(2))) __bf16 bf16x2;
static __device__ __forceinline__ uint32_t pack_bf(float lo, float hi) {
  bf16x2 v = __builtin_amdgcn_cvt_pk_bf16_f32(lo, hi);
  return __builtin_bit_cast(uint32_t, v);
}
static __device__ __forceinline__ uint32_t f2bf(float f) {
  return pack_bf(f, 0.0f) & 0xffffu;
}
#else
static __device__ __forceinline__ uint32_t f2bf(float f) {
  uint32_t u = __builtin_bit_cast(uint32_t, f);
  u += 0x7FFFu + ((u >> 16) & 1u);
  return u >> 16;
}
static __device__ __forceinline__ uint32_t pack_bf(float lo, float hi) {
  return f2bf(lo) | (f2bf(hi) << 16);
}
#endif

static __device__ __forceinline__ v8f wmma_bf16(u32x8 a, u32x8 b, v8f c) {
  return __builtin_amdgcn_wmma_f32_16x16x32_bf16(
      false, __builtin_bit_cast(bf16x16, a),
      false, __builtin_bit_cast(bf16x16, b),
      (short)0, c, false, false);
}
// 16-bit A-matrix 16x32 K offset for VGPR r, lane-half h (contiguous order):
//   lanes0-15 : r0..7 -> K=0..15
//   lanes16-31: r0..7 -> K=16..31
static __device__ __forceinline__ int a_kof(int r, int half) {
  return 2 * r + half * 16;
}

// ---- elementwise f32 -> bf16 ----------------------------------------------
__global__ void cvt_bf16_kernel(const float* __restrict__ in,
                                unsigned short* __restrict__ out, int n) {
  int i = blockIdx.x * blockDim.x + threadIdx.x;
  if (i < n) out[i] = (unsigned short)f2bf(in[i]);
}

// ---- transpose + convert a [K x N] f32 weight to [N x K] bf16 -------------
__global__ void cvt_t_kernel(const float* __restrict__ in,
                             unsigned short* __restrict__ out, int K, int N) {
  int i = blockIdx.x * blockDim.x + threadIdx.x;
  if (i < K * N) {
    int k = i / N, n = i % N;
    out[n * K + k] = (unsigned short)f2bf(in[i]);
  }
}

// ---- kernel 1: projections  out = a1@W1 (+ a2@W2), all-bf16 operands ------
// W*t are transposed weights: Wt[col*EE + k].
// VLAYOUT==0: out[((b*H+h)*rows + row)*16 + d]   (Q, K)
// VLAYOUT==1: out[((b*H+h)*16 + d)*rows + row]   (V, transposed for PV B-tiles)
template <bool HAS2, int VLAYOUT>
static __device__ __forceinline__ void proj_body(
    const unsigned short* __restrict__ a1g, const unsigned short* __restrict__ a2g,
    const unsigned short* __restrict__ W1t, const unsigned short* __restrict__ W2t,
    unsigned short* __restrict__ out, int rows) {
  const int lane = threadIdx.x, half = lane >> 4, l16 = lane & 15;
  const int h    = blockIdx.x;        // one head == one 16-col tile of HD
  const int row0 = blockIdx.y * 16;
  const int b    = blockIdx.z;
  const unsigned short* a1 = a1g + (size_t)b * rows * EE + (row0 + l16) * EE;
  const unsigned short* a2 = HAS2 ? a2g + (size_t)b * rows * EE + (row0 + l16) * EE
                                  : nullptr;
  const int col = h * 16 + l16;
  v8f acc = {};
#pragma unroll
  for (int k0 = 0; k0 < EE; k0 += 32) {
    u32x8 av, bv;
#pragma unroll
    for (int r = 0; r < 8; ++r) {
      av[r] = *(const uint32_t*)(a1 + k0 + a_kof(r, half));
      bv[r] = *(const uint32_t*)(W1t + (size_t)col * EE + k0 + 2 * r + half * 16);
    }
    acc = wmma_bf16(av, bv, acc);
    if (HAS2) {
      u32x8 av2, bv2;
#pragma unroll
      for (int r = 0; r < 8; ++r) {
        av2[r] = *(const uint32_t*)(a2 + k0 + a_kof(r, half));
        bv2[r] = *(const uint32_t*)(W2t + (size_t)col * EE + k0 + 2 * r + half * 16);
      }
      acc = wmma_bf16(av2, bv2, acc);
    }
  }
#pragma unroll
  for (int j = 0; j < 8; ++j) {
    const int row = row0 + j + 8 * half;           // C-layout: M = j + 8*half
    size_t idx = VLAYOUT
        ? (((size_t)b * HH + h) * DD + l16) * rows + row
        : (((size_t)b * HH + h) * rows + row) * DD + l16;
    out[idx] = (unsigned short)f2bf(acc[j]);
  }
}

__global__ __launch_bounds__(32) void proj_q_kernel(
    const unsigned short* __restrict__ a1, const unsigned short* __restrict__ a2,
    const unsigned short* __restrict__ W1t, const unsigned short* __restrict__ W2t,
    unsigned short* __restrict__ out, int rows) {
  proj_body<true, 0>(a1, a2, W1t, W2t, out, rows);
}
__global__ __launch_bounds__(32) void proj_k_kernel(
    const unsigned short* __restrict__ a1, const unsigned short* __restrict__ W1t,
    unsigned short* __restrict__ out, int rows) {
  proj_body<false, 0>(a1, nullptr, W1t, nullptr, out, rows);
}
__global__ __launch_bounds__(32) void proj_v_kernel(
    const unsigned short* __restrict__ a1, const unsigned short* __restrict__ W1t,
    unsigned short* __restrict__ out, int rows) {
  proj_body<false, 1>(a1, nullptr, W1t, nullptr, out, rows);
}

// ---- kernel 2: flash attention per (b, h, 16-query tile) -------------------
__global__ __launch_bounds__(32) void attn_kernel(
    const unsigned short* __restrict__ Qw, const unsigned short* __restrict__ Kw,
    const unsigned short* __restrict__ Vw, const float* __restrict__ mask,
    unsigned short* __restrict__ Ow) {
  __shared__ unsigned short Pl[16 * 32];     // P tile (16 g x 32 n), bf16
  __shared__ unsigned short Ktile[32 * 16];  // K tile (32 n x 16 d), bf16
  const int lane = threadIdx.x, half = lane >> 4, l16 = lane & 15;
  const int g0 = blockIdx.x * 16, h = blockIdx.y, b = blockIdx.z;
  const size_t bh = (size_t)b * HH + h;

  // Q as A-matrix, K-dim 16 zero-padded to 32 (lanes16-31 = 0)
  u32x8 qa;
#pragma unroll
  for (int r = 0; r < 8; ++r) {
    if (half == 0) {
      const int d = 2 * r;
      qa[r] = *(const uint32_t*)(Qw + (bh * GG + g0 + l16) * DD + d);
    } else qa[r] = 0u;
  }

  v8f acc = {};
  float mrow[8], lrow[8];
#pragma unroll
  for (int j = 0; j < 8; ++j) { mrow[j] = -3.0e38f; lrow[j] = 0.0f; }
  const float* mbase = mask + (size_t)b * GG * NN;
  const unsigned short* vrow = Vw + (bh * DD + l16) * NN + half * 16;
  const unsigned kof = (unsigned)(uintptr_t)(&Ktile[0]) + lane * 32;  // LDS byte addr

  for (int n0 = 0; n0 < NN; n0 += 32) {
    // --- async-stage K tile: lane copies row n0+lane (32B) into LDS ---------
    const unsigned long long ga =
        (unsigned long long)(Kw + (bh * NN + n0 + lane) * DD);
    asm volatile("s_wait_dscnt 0x0" ::: "memory");  // LDS reuse guard
    asm volatile(
        "global_load_async_to_lds_b128 %0, %1, off\n\t"
        "global_load_async_to_lds_b128 %0, %1, off offset:16"
        :: "v"(kof), "v"(ga) : "memory");

    // overlap the async copy with the mask loads for this 16x32 tile
    float ma0[8], ma1[8];
#pragma unroll
    for (int j = 0; j < 8; ++j) {
      const int g = g0 + j + 8 * half;
      ma0[j] = mbase[(size_t)g * NN + n0 + l16];
      ma1[j] = mbase[(size_t)g * NN + n0 + 16 + l16];
    }
    asm volatile("s_wait_asynccnt 0x0" ::: "memory");

    // K^T as B-matrix from LDS (d is the K-dim; lanes16-31 = zero padding)
    u32x8 kb0, kb1;
#pragma unroll
    for (int r = 0; r < 8; ++r) {
      if (half == 0) {
        kb0[r] = *(const uint32_t*)((const char*)Ktile + l16 * 32 + 4 * r);
        kb1[r] = *(const uint32_t*)((const char*)Ktile + (16 + l16) * 32 + 4 * r);
      } else { kb0[r] = 0u; kb1[r] = 0u; }
    }
    v8f zc = {};
    v8f s0 = wmma_bf16(qa, kb0, zc);
    v8f s1 = wmma_bf16(qa, kb1, zc);

#pragma unroll
    for (int j = 0; j < 8; ++j) {
      float a0 = s0[j] * 0.25f + ma0[j];
      float a1 = s1[j] * 0.25f + ma1[j];
      // row max across the 16-lane group (matches C-layout row mapping)
      float t = fmaxf(a0, a1);
      t = fmaxf(t, __shfl_xor(t, 1));
      t = fmaxf(t, __shfl_xor(t, 2));
      t = fmaxf(t, __shfl_xor(t, 4));
      t = fmaxf(t, __shfl_xor(t, 8));
      const float mnew  = fmaxf(mrow[j], t);
      const float scale = __expf(mrow[j] - mnew);
      const float p0 = __expf(a0 - mnew);
      const float p1 = __expf(a1 - mnew);
      float rs = p0 + p1;
      rs += __shfl_xor(rs, 1);
      rs += __shfl_xor(rs, 2);
      rs += __shfl_xor(rs, 4);
      rs += __shfl_xor(rs, 8);
      lrow[j] = lrow[j] * scale + rs;
      mrow[j] = mnew;
      acc[j] *= scale;
      Pl[(j + 8 * half) * 32 +      l16] = (unsigned short)f2bf(p0);
      Pl[(j + 8 * half) * 32 + 16 + l16] = (unsigned short)f2bf(p1);
    }
    __syncthreads();
    // P (A-layout via LDS) x V (B-layout, contiguous n pairs from [b,h,d,n])
    u32x8 pa, vb;
#pragma unroll
    for (int r = 0; r < 8; ++r) {
      pa[r] = *(const uint32_t*)(Pl + l16 * 32 + a_kof(r, half));
      vb[r] = *(const uint32_t*)(vrow + n0 + 2 * r);
    }
    acc = wmma_bf16(pa, vb, acc);
    __syncthreads();
  }
#pragma unroll
  for (int j = 0; j < 8; ++j) {
    const int g = g0 + j + 8 * half;
    const float inv = __builtin_amdgcn_rcpf(lrow[j]);
    Ow[((size_t)b * GG + g) * HDIM + h * DD + l16] =
        (unsigned short)f2bf(acc[j] * inv);
  }
}

// ---- kernel 3: mh = out_concat @ W_comb + b_comb (Wct transposed bf16) ----
__global__ __launch_bounds__(32) void comb_kernel(
    const unsigned short* __restrict__ Ow, const unsigned short* __restrict__ Wct,
    const float* __restrict__ bias, unsigned short* __restrict__ MH) {
  const int lane = threadIdx.x, half = lane >> 4, l16 = lane & 15;
  const int e0 = blockIdx.x * 16, row0 = blockIdx.y * 16, b = blockIdx.z;
  v8f acc = {};
#pragma unroll
  for (int k0 = 0; k0 < HDIM; k0 += 32) {
    u32x8 av, bv;
#pragma unroll
    for (int r = 0; r < 8; ++r) {
      av[r] = *(const uint32_t*)(Ow + ((size_t)b * GG + row0 + l16) * HDIM +
                                 k0 + a_kof(r, half));
      bv[r] = *(const uint32_t*)(Wct + (size_t)(e0 + l16) * HDIM +
                                 k0 + 2 * r + half * 16);
    }
    acc = wmma_bf16(av, bv, acc);
  }
  const float bs = bias[e0 + l16];
#pragma unroll
  for (int j = 0; j < 8; ++j) {
    const int row = row0 + j + 8 * half;
    MH[((size_t)b * GG + row) * EE + e0 + l16] = (unsigned short)f2bf(acc[j] + bs);
  }
}

// ---- kernel 4: probs = softmax(10*tanh(mh @ nodes^T / sqrtE) + mask) ------
__global__ __launch_bounds__(32) void final_kernel(
    const unsigned short* __restrict__ MH, const unsigned short* __restrict__ NB,
    const float* __restrict__ mask, float* __restrict__ out) {
  __shared__ float S[16 * NN];   // 32 KB score tile
  const int lane = threadIdx.x, half = lane >> 4, l16 = lane & 15;
  const int g0 = blockIdx.x * 16, b = blockIdx.y;
  // A = mh tile, all 4 K-chunks resident (32 VGPR-words)
  u32x8 av[4];
#pragma unroll
  for (int kc = 0; kc < 4; ++kc)
#pragma unroll
    for (int r = 0; r < 8; ++r)
      av[kc][r] = *(const uint32_t*)(MH + ((size_t)b * GG + g0 + l16) * EE +
                                     kc * 32 + a_kof(r, half));
  float mrow[8];
#pragma unroll
  for (int j = 0; j < 8; ++j) mrow[j] = -3.0e38f;
  const float* mk = mask + (size_t)b * GG * NN;
  const float inv_sqrt_e = 1.0f / 11.313708498984761f;

  for (int n0 = 0; n0 < NN; n0 += 16) {
    v8f acc = {};
#pragma unroll
    for (int kc = 0; kc < 4; ++kc) {
      u32x8 bv;
#pragma unroll
      for (int r = 0; r < 8; ++r)
        bv[r] = *(const uint32_t*)(NB + ((size_t)b * NN + n0 + l16) * EE +
                                   kc * 32 + 2 * r + half * 16);
      acc = wmma_bf16(av[kc], bv, acc);
    }
#pragma unroll
    for (int j = 0; j < 8; ++j) {
      const int g = g0 + j + 8 * half;
      // fast tanh: clamp then (e^2x - 1)/(e^2x + 1) via v_exp/v_rcp
      float x = acc[j] * inv_sqrt_e;
      x = fminf(fmaxf(x, -15.0f), 15.0f);
      const float e2 = __expf(2.0f * x);
      const float th = (e2 - 1.0f) * __builtin_amdgcn_rcpf(e2 + 1.0f);
      float s = 10.0f * th + mk[(size_t)g * NN + n0 + l16];
      S[(j + 8 * half) * NN + n0 + l16] = s;
      float t = s;
      t = fmaxf(t, __shfl_xor(t, 1));
      t = fmaxf(t, __shfl_xor(t, 2));
      t = fmaxf(t, __shfl_xor(t, 4));
      t = fmaxf(t, __shfl_xor(t, 8));
      mrow[j] = fmaxf(mrow[j], t);
    }
  }
  __syncthreads();
  float lrow[8];
#pragma unroll
  for (int j = 0; j < 8; ++j) lrow[j] = 0.0f;
  for (int n0 = 0; n0 < NN; n0 += 16)
#pragma unroll
    for (int j = 0; j < 8; ++j)
      lrow[j] += __expf(S[(j + 8 * half) * NN + n0 + l16] - mrow[j]);
#pragma unroll
  for (int j = 0; j < 8; ++j) {
    float t = lrow[j];
    t += __shfl_xor(t, 1);
    t += __shfl_xor(t, 2);
    t += __shfl_xor(t, 4);
    t += __shfl_xor(t, 8);
    lrow[j] = __builtin_amdgcn_rcpf(t);
  }
  for (int n0 = 0; n0 < NN; n0 += 16)
#pragma unroll
    for (int j = 0; j < 8; ++j) {
      const int g = g0 + j + 8 * half;
      out[((size_t)b * GG + g) * NN + n0 + l16] =
          __expf(S[(j + 8 * half) * NN + n0 + l16] - mrow[j]) * lrow[j];
    }
}

// ---- host side -------------------------------------------------------------
extern "C" void kernel_launch(void* const* d_in, const int* in_sizes, int n_in,
                              void* d_out, int out_size, void* d_ws, size_t ws_size,
                              hipStream_t stream) {
  (void)in_sizes; (void)n_in; (void)out_size; (void)ws_size;
  const float* nodes = (const float*)d_in[0];
  const float* q1    = (const float*)d_in[1];
  const float* last  = (const float*)d_in[2];
  const float* mask  = (const float*)d_in[3];
  const float* Wqf   = (const float*)d_in[4];
  const float* Wql   = (const float*)d_in[5];
  const float* Wk    = (const float*)d_in[6];
  const float* Wv    = (const float*)d_in[7];
  const float* Wc    = (const float*)d_in[8];
  const float* bc    = (const float*)d_in[9];
  float* out = (float*)d_out;

  // workspace: eight 4 MB bf16 tensors + five 32 KB bf16 weights (~32.2 MB)
  const size_t S1 = (size_t)BB * GG * HDIM * sizeof(unsigned short);
  const size_t SW = (size_t)EE * HDIM * sizeof(unsigned short);
  char* w = (char*)d_ws;
  unsigned short* Qw   = (unsigned short*)(w + 0 * S1);
  unsigned short* Kw   = (unsigned short*)(w + 1 * S1);
  unsigned short* Vw   = (unsigned short*)(w + 2 * S1);
  unsigned short* Ow   = (unsigned short*)(w + 3 * S1);
  unsigned short* MH   = (unsigned short*)(w + 4 * S1);
  unsigned short* NB   = (unsigned short*)(w + 5 * S1);
  unsigned short* Q1b  = (unsigned short*)(w + 6 * S1);
  unsigned short* Lsb  = (unsigned short*)(w + 7 * S1);
  unsigned short* Wqft = (unsigned short*)(w + 8 * S1 + 0 * SW);
  unsigned short* Wqlt = (unsigned short*)(w + 8 * S1 + 1 * SW);
  unsigned short* Wkt  = (unsigned short*)(w + 8 * S1 + 2 * SW);
  unsigned short* Wvt  = (unsigned short*)(w + 8 * S1 + 3 * SW);
  unsigned short* Wct  = (unsigned short*)(w + 8 * S1 + 4 * SW);

  const int nact = BB * GG * EE;
  hipLaunchKernelGGL(cvt_bf16_kernel, dim3((nact + 255) / 256), dim3(256), 0,
                     stream, nodes, NB, nact);
  hipLaunchKernelGGL(cvt_bf16_kernel, dim3((nact + 255) / 256), dim3(256), 0,
                     stream, q1, Q1b, nact);
  hipLaunchKernelGGL(cvt_bf16_kernel, dim3((nact + 255) / 256), dim3(256), 0,
                     stream, last, Lsb, nact);
  const int nw = EE * HDIM;
  dim3 wg((nw + 255) / 256), wb(256);
  hipLaunchKernelGGL(cvt_t_kernel, wg, wb, 0, stream, Wqf, Wqft, EE, HDIM);
  hipLaunchKernelGGL(cvt_t_kernel, wg, wb, 0, stream, Wql, Wqlt, EE, HDIM);
  hipLaunchKernelGGL(cvt_t_kernel, wg, wb, 0, stream, Wk,  Wkt,  EE, HDIM);
  hipLaunchKernelGGL(cvt_t_kernel, wg, wb, 0, stream, Wv,  Wvt,  EE, HDIM);
  hipLaunchKernelGGL(cvt_t_kernel, wg, wb, 0, stream, Wc,  Wct,  HDIM, EE);

  dim3 pg(HDIM / 16, GG / 16, BB), pb(32);
  hipLaunchKernelGGL(proj_q_kernel, pg, pb, 0, stream, Q1b, Lsb, Wqft, Wqlt, Qw, GG);
  hipLaunchKernelGGL(proj_k_kernel, pg, pb, 0, stream, NB, Wkt, Kw, NN);
  hipLaunchKernelGGL(proj_v_kernel, pg, pb, 0, stream, NB, Wvt, Vw, NN);
  hipLaunchKernelGGL(attn_kernel, dim3(GG / 16, HH, BB), dim3(32), 0, stream,
                     Qw, Kw, Vw, mask, Ow);
  hipLaunchKernelGGL(comb_kernel, dim3(EE / 16, GG / 16, BB), dim3(32), 0, stream,
                     Ow, Wct, bc, MH);
  hipLaunchKernelGGL(final_kernel, dim3(GG / 16, BB), dim3(32), 0, stream,
                     MH, NB, mask, out);
}
